// oklamai_LSTM_66709432041647
// MI455X (gfx1250) — matrix operand order repaired and checked by reading, not compiled, not run on hardware
//
#include <hip/hip_runtime.h>
#include <hip/hip_bf16.h>
#include <math.h>

// LSTM: SEQ=4096 timesteps, IN=1024, HID=1024, gates=4*HID=4096.
//
// Phase 1: pre = x @ W_ih.T + (b_ih+b_hh)  -> bf16 WMMA GEMM, fp32 accum.
// Phase 2: persistent scan, 32 WGs. WG b owns hidden units [b*32,b*32+32)
//          and the 128 matching W_hh rows (4 gates) resident in LDS as bf16
//          (256 KB of the 320 KB/WGP budget). Per step, each wave computes
//          16 gate rows with TWO independent chains of 16 chained
//          v_wmma_f32_16x16x32_bf16 (even/odd K tiles) for deeper matrix-
//          pipe ILP on the latency-bound sequential path:
//            A = h broadcast into all 16 rows (lane-replicated loads)
//            B = W_hh^T tile streamed from LDS (contiguous ds_load_b128)
//            C(chain0) seeded with pre[t] slice  => no reduction needed.
//          h_t is exchanged through d_out with a per-timestep
//          release/acquire counter barrier in d_ws.

#define SEQ 4096
#define IN  1024
#define HID 1024
#define G4  (4 * HID)   // 4096
#define NBLK 32         // scan workgroups; 32 * 256KB LDS == all of W_hh (bf16)
#define UPB  (HID / NBLK)      // 32 hidden units per block
#define ROWS (4 * UPB)         // 128 W_hh rows per block

typedef __attribute__((ext_vector_type(16))) __bf16 v16bf;
typedef __attribute__((ext_vector_type(8)))  __bf16 v8bf;
typedef __attribute__((ext_vector_type(8)))  float  v8f;

// ---------------------------------------------------------------- init
__global__ void lstm_init_counters(unsigned int* __restrict__ cnt) {
    int i = blockIdx.x * blockDim.x + threadIdx.x;
    if (i < SEQ) cnt[i] = 0u;
}

// ---------------------------------------------------------------- phase 1
// Block: 256 threads = 8 waves; each wave computes one 16x16 tile of pre.
// grid = (G4/(8*16), SEQ/16) = (32, 256).
__global__ __launch_bounds__(256)
void lstm_pre_gemm(const float* __restrict__ x,      // (SEQ, IN)
                   const float* __restrict__ W_ih,   // (G4, IN)
                   const float* __restrict__ b_ih,   // (G4)
                   const float* __restrict__ b_hh,   // (G4)
                   float* __restrict__ pre)          // (SEQ, G4)
{
    const int lane = threadIdx.x & 31;
    const int wave = threadIdx.x >> 5;
    const int m0 = blockIdx.y * 16;                  // row tile (time)
    const int n0 = (blockIdx.x * 8 + wave) * 16;     // col tile (gate dim)

    const int mrow = m0 + (lane & 15);
    const int ncol = n0 + (lane & 15);
    const int kbA  = (lane >> 4) ? 16 : 0;           // A: contiguous K half
    const int khB  = (lane >> 4) ? 16 : 0;           // B: contiguous K half

    const float* __restrict__ xrow = x    + (size_t)mrow * IN;
    const float* __restrict__ wrow = W_ih + (size_t)ncol * IN;

    v8f acc = {};
    for (int k0 = 0; k0 < IN; k0 += 32) {
        v16bf a, b;
        // A 16x32: lane l -> row (l&15), K = kbA..kbA+15 contiguous
        #pragma unroll
        for (int j = 0; j < 8; ++j) {
            a[j]     = (__bf16)xrow[k0 + kbA + j];
            a[8 + j] = (__bf16)xrow[k0 + kbA + 8 + j];
        }
        // B 32x16: lane l -> column (l&15), K = khB..khB+15 contiguous
        #pragma unroll
        for (int j = 0; j < 16; ++j)
            b[j] = (__bf16)wrow[k0 + khB + j];
        acc = __builtin_amdgcn_wmma_f32_16x16x32_bf16(
                  false, a, false, b, (short)0, acc, false, false);
    }

    // C/D layout: VGPR j -> row m0 + j + 8*(lane>=16), col n0 + (lane&15).
    const float bias  = b_ih[ncol] + b_hh[ncol];
    const int   mbase = m0 + ((lane >> 4) ? 8 : 0);
    #pragma unroll
    for (int j = 0; j < 8; ++j)
        pre[(size_t)(mbase + j) * G4 + ncol] = acc[j] + bias;
}

// ---------------------------------------------------------------- phase 2
__device__ __forceinline__ float sigmoidf_(float v) {
    return 1.0f / (1.0f + __expf(-v));
}

// grid = NBLK blocks, 256 threads (8 waves). Dynamic LDS:
//   hb_s: 1024 bf16  (2048 B)   h_{t-1}, staged each step
//   g_s :  128 f32   ( 512 B)   finished gate pre-activations
//   c_s :   32 f32   ( 128 B)   cell state (lives across all 4096 steps)
//   w_s : 128x1024 bf16 (262144 B) resident W_hh slice
__global__ __launch_bounds__(256)
void lstm_scan(const float* __restrict__ W_hh,       // (G4, HID)
               const float* __restrict__ pre,        // (SEQ, G4)
               float* __restrict__ hout,             // (SEQ, HID) == d_out
               unsigned int* __restrict__ cnt)       // (SEQ)
{
    extern __shared__ char smem[];
    __bf16* hb_s = (__bf16*)smem;                     // 1024 bf16
    float*  g_s  = (float*)(smem + 2048);             // 128
    float*  c_s  = g_s + ROWS;                        // 32
    __bf16* w_s  = (__bf16*)(c_s + UPB);              // 128*1024 (32B aligned)

    const int tid  = threadIdx.x;
    const int lane = tid & 31;
    const int wave = tid >> 5;
    const int ln   = lane & 15;
    const int kbA  = (lane >> 4) ? 16 : 0;            // A K-run base
    const int khB  = (lane >> 4) ? 16 : 0;            // B K-half base
    const int u0   = blockIdx.x * UPB;                // first hidden unit owned

    // ---- one-time: stage this block's W_hh slice into LDS as bf16 ----
    for (int idx = tid; idx < ROWS * HID; idx += 256) {
        const int r = idx >> 10;                      // 0..127 (gate*32 + u)
        const int k = idx & (HID - 1);
        const int grow = (r >> 5) * HID + u0 + (r & (UPB - 1));
        w_s[idx] = (__bf16)W_hh[(size_t)grow * HID + k];
    }
    if (tid < UPB) c_s[tid] = 0.0f;
    __syncthreads();

    // This wave owns block-local gate rows [wave*16, wave*16+16).
    const int rloc  = wave * 16 + ln;                 // 0..127
    const int grow  = (rloc >> 5) * HID + u0 + (rloc & (UPB - 1));
    // B fragment base: column ln = W_hh row rloc, contiguous K from LDS.
    const __bf16* wfrag = &w_s[(size_t)rloc * HID + khB];

    for (int t = 0; t < SEQ; ++t) {
        // fetch h_{t-1} (release/acquire barrier across the 32 WGs)
        if (t > 0) {
            while (__hip_atomic_load(&cnt[t - 1], __ATOMIC_ACQUIRE,
                                     __HIP_MEMORY_SCOPE_AGENT) < NBLK)
                __builtin_amdgcn_s_sleep(2);
            const float* hprev = hout + (size_t)(t - 1) * HID;
            for (int i = tid; i < HID; i += 256)
                hb_s[i] = (__bf16)hprev[i];           // v_cvt_pk_bf16_f32
        }
        // chain-0 accumulator seeded with pre[t] (bias already folded in):
        // only D row 0 matters; it lives in VGPR0 lanes 0..15.
        v8f acc0 = {}, acc1 = {};
        const float pv = pre[(size_t)t * G4 + grow];
        acc0[0] = (lane < 16) ? pv : 0.0f;
        if (t + 1 < SEQ)                              // global_prefetch_b8
            __builtin_prefetch(&pre[(size_t)(t + 1) * G4 + grow], 0, 1);
        __syncthreads();

        if (t > 0) {
            // Two independent 16-deep WMMA chains over even/odd K tiles.
            #pragma unroll 2
            for (int k0 = 0; k0 < HID; k0 += 64) {
                // A: every lane replicates h -> all 16 A rows equal h.
                const v8bf a0e = *(const v8bf*)&hb_s[k0 + kbA];
                const v8bf a1e = *(const v8bf*)&hb_s[k0 + kbA + 8];
                const v16bf ae = __builtin_shufflevector(
                    a0e, a1e, 0, 1, 2, 3, 4, 5, 6, 7,
                    8, 9, 10, 11, 12, 13, 14, 15);
                const v16bf be = *(const v16bf*)&wfrag[k0];
                acc0 = __builtin_amdgcn_wmma_f32_16x16x32_bf16(
                           false, ae, false, be, (short)0, acc0, false, false);

                const v8bf a0o = *(const v8bf*)&hb_s[k0 + 32 + kbA];
                const v8bf a1o = *(const v8bf*)&hb_s[k0 + 32 + kbA + 8];
                const v16bf ao = __builtin_shufflevector(
                    a0o, a1o, 0, 1, 2, 3, 4, 5, 6, 7,
                    8, 9, 10, 11, 12, 13, 14, 15);
                const v16bf bo = *(const v16bf*)&wfrag[k0 + 32];
                acc1 = __builtin_amdgcn_wmma_f32_16x16x32_bf16(
                           false, ao, false, bo, (short)0, acc1, false, false);
            }
        }
        if (lane < 16) g_s[rloc] = acc0[0] + acc1[0]; // finished gate row
        __syncthreads();

        if (tid < UPB) {
            const float gi = sigmoidf_(g_s[tid]);
            const float gf = sigmoidf_(g_s[UPB + tid]);
            const float gg = tanhf(g_s[2 * UPB + tid]);
            const float go = sigmoidf_(g_s[3 * UPB + tid]);
            const float cn = gf * c_s[tid] + gi * gg;
            c_s[tid] = cn;
            const float hn = go * tanhf(cn);
            __hip_atomic_store(&hout[(size_t)t * HID + u0 + tid], hn,
                               __ATOMIC_RELAXED, __HIP_MEMORY_SCOPE_AGENT);
        }
        __syncthreads();
        if (tid == 0)
            __hip_atomic_fetch_add(&cnt[t], 1u, __ATOMIC_RELEASE,
                                   __HIP_MEMORY_SCOPE_AGENT);
    }
}

// ---------------------------------------------------------------- launch
extern "C" void kernel_launch(void* const* d_in, const int* in_sizes, int n_in,
                              void* d_out, int out_size, void* d_ws, size_t ws_size,
                              hipStream_t stream) {
    const float* x    = (const float*)d_in[0];   // (SEQ,1,IN)
    const float* W_ih = (const float*)d_in[1];   // (G4,IN)
    const float* W_hh = (const float*)d_in[2];   // (G4,HID)
    const float* b_ih = (const float*)d_in[3];   // (G4)
    const float* b_hh = (const float*)d_in[4];   // (G4)
    float*       hout = (float*)d_out;           // (SEQ,1,HID)

    // workspace: [0,64KB) timestep counters; [64KB, 64KB+64MB) pre
    unsigned int* cnt = (unsigned int*)d_ws;
    float*        pre = (float*)((char*)d_ws + (64u << 10));

    lstm_init_counters<<<(SEQ + 255) / 256, 256, 0, stream>>>(cnt);

    dim3 ggrid(G4 / (8 * 16), SEQ / 16);         // (32, 256)
    lstm_pre_gemm<<<ggrid, 256, 0, stream>>>(x, W_ih, b_ih, b_hh, pre);

    const size_t smem = 2048                     // hb_s (bf16 h)
                      + ROWS * 4 + UPB * 4       // g_s, c_s
                      + (size_t)ROWS * HID * 2;  // w_s: 256 KB (of 320 KB/WGP)
    static bool attr_set = false;                // idempotent host-side attr
    if (!attr_set) {
        hipFuncSetAttribute((const void*)lstm_scan,
                            hipFuncAttributeMaxDynamicSharedMemorySize,
                            (int)smem);
        attr_set = true;
    }
    lstm_scan<<<NBLK, 256, smem, stream>>>(W_hh, pre, hout, cnt);
}
